// Attention_27865747817251
// MI455X (gfx1250) — matrix operand order repaired and checked by reading, not compiled, not run on hardware
//
#include <hip/hip_runtime.h>
#include <stdint.h>

// Problem dims (match the reference).
constexpr int Bb  = 2;
constexpr int Ll  = 2048;
constexpr int Dd  = 2048;
constexpr int NHh = 16;
constexpr int HDd = 128;   // D / NH

typedef __attribute__((ext_vector_type(16))) _Float16 v16h;
typedef __attribute__((ext_vector_type(8)))  _Float16 v8h;
typedef __attribute__((ext_vector_type(4)))  _Float16 v4h;
typedef __attribute__((ext_vector_type(8)))  float    v8f;

static __device__ __forceinline__ v16h cat8(v8h lo, v8h hi) {
  return __builtin_shufflevector(lo, hi, 0,1,2,3,4,5,6,7,8,9,10,11,12,13,14,15);
}
static __device__ __forceinline__ v8f wmma_f16(v16h a, v16h b, v8f c) {
  // D = A(16x32 f16) x B(32x16 f16) + C(16x16 f32)
  return __builtin_amdgcn_wmma_f32_16x16x32_f16(false, a, false, b, (short)0, c,
                                                false, false);
}

// ---- gfx1250 async global->LDS copy via inline asm (ASYNCcnt path) --------
// The ROCm clang builtin for this op has a toolchain-specific prototype, so we
// emit the ISA instruction directly (GV addressing: 64-bit vaddr, saddr=off).
// LDS destination address = low 32 bits of the generic shared pointer.
static __device__ __forceinline__ void async_ld_lds_b128(_Float16* dst,
                                                         const _Float16* src) {
  const uint32_t lds = (uint32_t)(uintptr_t)dst;
  const uint64_t gad = (uint64_t)(uintptr_t)src;
  asm volatile("global_load_async_to_lds_b128 %0, %1, off"
               :: "v"(lds), "v"(gad) : "memory");
}
static __device__ __forceinline__ void wait_async0() {
  asm volatile("s_wait_asynccnt 0x0" ::: "memory");
}

// Copy one 32x128 f16 chunk (8 KB) global -> LDS, 64 threads cooperating.
static __device__ __forceinline__ void copy_chunk_8k(_Float16* dst,
                                                     const _Float16* src,
                                                     int tid) {
  #pragma unroll
  for (int c = tid; c < 512; c += 64)
    async_ld_lds_b128(dst + c * 8, src + c * 8);
}

// ---------------------------------------------------------------------------
// Generic WMMA GEMM:  C[M,N] = A[M,K] (row-major) * B[K,N] (row-major) + bias
// A element type AT is float or _Float16 (converted to f16 while staging LDS).
// Block tile 128x128, 256 threads (8 waves), each wave computes 32x64.
// ---------------------------------------------------------------------------
template <typename AT, bool OUT_HALF>
__global__ void __launch_bounds__(256)
gemm_wmma(const AT* __restrict__ A, const float* __restrict__ B,
          const float* __restrict__ bias, void* __restrict__ Cout,
          int M, int N, int K)
{
  __shared__ __align__(32) _Float16 As[128 * 32];   // [m][k]
  __shared__ __align__(32) _Float16 Bt[128 * 32];   // [n][k]  (transposed)

  const int tid  = threadIdx.x;
  const int lane = tid & 31;
  const int w    = tid >> 5;
  const int l16  = lane & 15;
  const int hf   = lane >> 4;

  const int m0 = blockIdx.y * 128;
  const int n0 = blockIdx.x * 128;
  const int mw = (w & 3) * 32;     // wave M offset in block tile
  const int nw = (w >> 2) * 64;    // wave N offset in block tile

  v8f acc[2][4];
  #pragma unroll
  for (int mi = 0; mi < 2; ++mi)
    #pragma unroll
    for (int ni = 0; ni < 4; ++ni) { v8f z = {}; acc[mi][ni] = z; }

  for (int k0 = 0; k0 < K; k0 += 32) {
    __syncthreads();   // protect previous iteration's LDS reads

    // ---- stage A tile (128x32) into LDS, converting to f16 ----
    #pragma unroll 2
    for (int c = tid; c < 128 * 32 / 4; c += 256) {
      const int r = c >> 3, cb = (c & 7) * 4;
      const AT* ap = &A[(size_t)(m0 + r) * K + k0 + cb];
      v4h hh = { (_Float16)(float)ap[0], (_Float16)(float)ap[1],
                 (_Float16)(float)ap[2], (_Float16)(float)ap[3] };
      *(v4h*)&As[r * 32 + cb] = hh;
    }
    // ---- stage B tile (32x128) transposed into LDS ----
    #pragma unroll 2
    for (int c = tid; c < 32 * 128 / 4; c += 256) {
      const int kk = c >> 5, nb = (c & 31) * 4;
      const float* bp = &B[(size_t)(k0 + kk) * N + n0 + nb];
      #pragma unroll
      for (int p = 0; p < 4; ++p)
        Bt[(nb + p) * 32 + kk] = (_Float16)bp[p];
    }
    if (k0 + 32 < K) {   // gfx1250 global_prefetch_b8 for the next K tile
      __builtin_prefetch(&A[(size_t)(m0 + (tid >> 1)) * K + k0 + 32], 0, 1);
      __builtin_prefetch(&B[(size_t)(k0 + 32 + w * 4) * N + n0 + lane * 4], 0, 1);
    }
    __syncthreads();

    // ---- A fragments: lane%16 = row; lane/16 selects K {0..15} vs {16..31}
    v16h afrag[2];
    #pragma unroll
    for (int mi = 0; mi < 2; ++mi) {
      const _Float16* p = &As[(mw + mi * 16 + l16) * 32 + hf * 16];
      afrag[mi] = cat8(*(const v8h*)p, *(const v8h*)(p + 8));
    }
    // ---- B fragments: lane%16 = column; K contiguous (16 halves) ----
    #pragma unroll
    for (int ni = 0; ni < 4; ++ni) {
      const v16h bfrag = *(const v16h*)&Bt[(nw + ni * 16 + l16) * 32 + hf * 16];
      #pragma unroll
      for (int mi = 0; mi < 2; ++mi)
        acc[mi][ni] = wmma_f16(afrag[mi], bfrag, acc[mi][ni]);
    }
  }

  // ---- epilogue: C fragment row = r + 8*(lane/16), col = lane%16 ----
  #pragma unroll
  for (int mi = 0; mi < 2; ++mi)
    #pragma unroll
    for (int ni = 0; ni < 4; ++ni) {
      const int n = n0 + nw + ni * 16 + l16;
      const float bn = bias[n];
      #pragma unroll
      for (int r = 0; r < 8; ++r) {
        const int m = m0 + mw + mi * 16 + r + 8 * hf;
        const float v = acc[mi][ni][r] + bn;
        if (OUT_HALF) ((_Float16*)Cout)[(size_t)m * N + n] = (_Float16)v;
        else          ((float*)Cout)[(size_t)m * N + n]    = v;
      }
    }
}

// ---------------------------------------------------------------------------
// Flash-style attention for the reference's (scrambled-reshape) layout.
// qkv: (B, L, 3D) row-major f16. Per batch, the flat reshape means
//   Q[h][i][d] = qkv[b*L*3D +          h*L*HD + i*HD + d]
//   K[h][i][d] = qkv[b*L*3D + 1*D*L  + h*L*HD + i*HD + d]
//   V[h][i][d] = qkv[b*L*3D + 2*D*L  + h*L*HD + i*HD + d]
// S = K . Q^T, softmax over j, O = P . V.
// 2 waves per block; wave w owns K-rows [i0+16w, i0+16w+16). The Q j-chunk is
// double-buffered in LDS and filled with async global->LDS copies pipelined
// one iteration ahead; the V j-chunk is staged transposed by both waves.
// Output written pre-transposed: attn[(b*L + i)*D + h*HD + d]  (f16).
// ---------------------------------------------------------------------------
__global__ void __launch_bounds__(64)
attn_fa(const _Float16* __restrict__ qkv, _Float16* __restrict__ attn)
{
  __shared__ __align__(32) _Float16 Qs[2][32 * HDd];  // double-buffered Q chunk (16 KB)
  __shared__ __align__(32) _Float16 Vt[HDd * 32];     // [d][j] transposed V chunk (8 KB)
  __shared__ __align__(32) _Float16 Pt[2][16 * 32];   // per-wave P tile (2 KB)

  const int tid  = threadIdx.x;
  const int lane = tid & 31;
  const int w    = tid >> 5;
  const int l16  = lane & 15;
  const int hf   = lane >> 4;
  const int i0   = blockIdx.x * 32 + w * 16;  // this wave's K-row tile
  const int h    = blockIdx.y;
  const int b    = blockIdx.z;

  const size_t bbase = (size_t)b * Ll * 3 * Dd;
  const _Float16* Qp = qkv + bbase + (size_t)h * Ll * HDd;
  const _Float16* Kp = Qp + (size_t)Dd * Ll;
  const _Float16* Vp = Kp + (size_t)Dd * Ll;

  // Persistent A fragments: K rows i0..i0+15, HD=128 -> 4 fragments of K=32.
  v16h afrag[4];
  #pragma unroll
  for (int kc = 0; kc < 4; ++kc) {
    const _Float16* p = Kp + (size_t)(i0 + l16) * HDd + kc * 32 + hf * 16;
    afrag[kc] = cat8(*(const v8h*)p, *(const v8h*)(p + 8));
  }

  v8f o[8];
  #pragma unroll
  for (int dn = 0; dn < 8; ++dn) { v8f z = {}; o[dn] = z; }
  float mrow[8], lrow[8];
  #pragma unroll
  for (int r = 0; r < 8; ++r) { mrow[r] = -3.0e38f; lrow[r] = 0.0f; }

  const float sscale = 0.08838834764831845f;   // 1/sqrt(128)

  // Prologue: start async copy of the first Q chunk.
  copy_chunk_8k(Qs[0], Qp, tid);

  for (int j0 = 0; j0 < Ll; j0 += 32) {
    const int cur = (j0 >> 5) & 1;
    wait_async0();     // Qs[cur] copy (issued last iteration) complete
    __syncthreads();   // all waves: copies visible, previous-iter reads done

    // Pipeline: kick off the next Q chunk into the other buffer.
    if (j0 + 32 < Ll)
      copy_chunk_8k(Qs[cur ^ 1], Qp + (size_t)(j0 + 32) * HDd, tid);

    // ---- stage V chunk (32 x 128) transposed into LDS (both waves) ----
    #pragma unroll 4
    for (int c = tid; c < 512; c += 64) {
      const int jj = c >> 4, dblk = (c & 15) * 8;
      const v8h vv = *(const v8h*)&Vp[(size_t)(j0 + jj) * HDd + dblk];
      #pragma unroll
      for (int p = 0; p < 8; ++p) Vt[(dblk + p) * 32 + jj] = vv[p];
    }

    // ---- S tile (16 x 32): two 16x16 WMMA outputs, K=128 in 4 steps ----
    v8f s[2];
    #pragma unroll
    for (int jn = 0; jn < 2; ++jn) {
      v8f sf = {};
      #pragma unroll
      for (int kc = 0; kc < 4; ++kc) {
        // B[k=d][n=j] = Q[j][d]: contiguous-in-d from the staged Q chunk.
        const v16h bfrag = *(const v16h*)
            &Qs[cur][(jn * 16 + l16) * HDd + kc * 32 + hf * 16];
        sf = wmma_f16(afrag[kc], bfrag, sf);
      }
      s[jn] = sf;
    }

    __syncthreads();   // Vt fully staged by both waves before V-fragment reads

    // ---- online softmax; write P (f16) to this wave's LDS tile ----
    #pragma unroll
    for (int r = 0; r < 8; ++r) {
      const float s0 = s[0][r] * sscale, s1 = s[1][r] * sscale;
      float mx = fmaxf(s0, s1);
      #pragma unroll
      for (int off = 8; off; off >>= 1) mx = fmaxf(mx, __shfl_xor(mx, off, 32));
      const float mn   = fmaxf(mrow[r], mx);
      const float corr = __expf(mrow[r] - mn);
      const float p0 = __expf(s0 - mn), p1 = __expf(s1 - mn);
      float rs = p0 + p1;
      #pragma unroll
      for (int off = 8; off; off >>= 1) rs += __shfl_xor(rs, off, 32);
      lrow[r] = lrow[r] * corr + rs;
      mrow[r] = mn;
      #pragma unroll
      for (int dn = 0; dn < 8; ++dn) o[dn][r] *= corr;
      const int m = r + 8 * hf;            // C-fragment row mapping
      Pt[w][m * 32 + l16]      = (_Float16)p0;
      Pt[w][m * 32 + 16 + l16] = (_Float16)p1;
    }

    // ---- P as A fragment (16x32), then O += P . V over 8 d-tiles ----
    // Pt[w] is written and read by the same wave: LDS ops are in-order.
    const _Float16* pp = &Pt[w][l16 * 32 + hf * 16];
    const v16h pfrag = cat8(*(const v8h*)pp, *(const v8h*)(pp + 8));
    #pragma unroll
    for (int dn = 0; dn < 8; ++dn) {
      const v16h vfrag = *(const v16h*)&Vt[(dn * 16 + l16) * 32 + hf * 16];
      o[dn] = wmma_f16(pfrag, vfrag, o[dn]);
    }
  }

  // ---- normalize and store, already transposed to (b, i, h*HD+d) ----
  #pragma unroll
  for (int r = 0; r < 8; ++r) {
    const int m = i0 + r + 8 * hf;
    const float inv = 1.0f / lrow[r];
    #pragma unroll
    for (int dn = 0; dn < 8; ++dn)
      attn[((size_t)(b * Ll + m)) * Dd + h * HDd + dn * 16 + l16] =
          (_Float16)(o[dn][r] * inv);
  }
}

// ---------------------------------------------------------------------------
// Launch: qkv GEMM (f16 out) -> attention -> output projection (f32 out).
// Workspace: qkv f16 (48 MB) + attn f16 (16 MB) = 64 MB in d_ws.
// ---------------------------------------------------------------------------
extern "C" void kernel_launch(void* const* d_in, const int* in_sizes, int n_in,
                              void* d_out, int out_size, void* d_ws, size_t ws_size,
                              hipStream_t stream) {
  const float* x     = (const float*)d_in[0];
  const float* w_qkv = (const float*)d_in[1];
  const float* b_qkv = (const float*)d_in[2];
  const float* w_out = (const float*)d_in[3];
  const float* b_out = (const float*)d_in[4];
  float* out = (float*)d_out;

  _Float16* qkv_h  = (_Float16*)d_ws;
  _Float16* attn_h = qkv_h + (size_t)Bb * Ll * 3 * Dd;

  const int M = Bb * Ll;          // 4096

  // qkv = x @ w_qkv + b_qkv   (M=4096, N=6144, K=2048), f16 output
  dim3 g1((3 * Dd) / 128, M / 128);
  gemm_wmma<float, true><<<g1, 256, 0, stream>>>(x, w_qkv, b_qkv, qkv_h,
                                                 M, 3 * Dd, Dd);

  // attention: 2 waves per block, each owning a 16-row K tile
  dim3 g2(Ll / 32, NHh, Bb);
  attn_fa<<<g2, 64, 0, stream>>>(qkv_h, attn_h);

  // out = attn @ w_out + b_out  (M=4096, N=2048, K=2048), f32 output
  dim3 g3(Dd / 128, M / 128);
  gemm_wmma<_Float16, false><<<g3, 256, 0, stream>>>(attn_h, w_out, b_out, out,
                                                     M, Dd, Dd);
}
